// backbone_msnet_7919919694162
// MI455X (gfx1250) — matrix operand order repaired and checked by reading, not compiled, not run on hardware
//
#include <hip/hip_runtime.h>
#include <hip/hip_bf16.h>

typedef __attribute__((ext_vector_type(16))) _Float16 v16h;
typedef __attribute__((ext_vector_type(8)))  _Float16 v8h;
typedef __attribute__((ext_vector_type(8)))  float    v8f;

union AFrag { v16h v; v8h h[2]; };

// ---------------------------------------------------------------------------
// Weight conversion: W (K x N, f32, row-major) -> WT (N x Kpad, f16, zero-pad)
// ---------------------------------------------------------------------------
__global__ void convert_wt_kernel(const float* __restrict__ W, _Float16* __restrict__ WT,
                                  int K, int Kpad, int N, int total)
{
    int t = blockIdx.x * blockDim.x + threadIdx.x;
    if (t >= total) return;
    int n = t / Kpad, k = t % Kpad;
    WT[t] = (k < K) ? (_Float16)W[(size_t)k * N + n] : (_Float16)0.0f;
}

// ---------------------------------------------------------------------------
// WMMA GEMM:  out = relu(A (MxK f16) @ WT^T (K x N) + bias)
// One wave per 16x32 output tile.  A row-major (stride K, K % 32 == 0),
// WT is N x K (each row is one output column of W).  Fragment loads follow the
// contiguous k = 16h+e order for A and B, all contiguous 128-bit loads.
// ---------------------------------------------------------------------------
__global__ void gemm_wmma_kernel(const _Float16* __restrict__ A,
                                 const _Float16* __restrict__ WT,
                                 const float* __restrict__ bias,
                                 _Float16* __restrict__ out_h,
                                 float* __restrict__ out_f,
                                 int M, int K, int N)
{
    int m0 = blockIdx.x * 16;
    int n0 = blockIdx.y * 32;
    int l  = threadIdx.x;
    int h  = l >> 4;      // lane half
    int r  = l & 15;

    v8f c0 = {}; v8f c1 = {};
    const _Float16* arow = A  + (size_t)(m0 + r) * K;
    const _Float16* b0p  = WT + (size_t)(n0 + r) * K;
    const _Float16* b1p  = WT + (size_t)(n0 + 16 + r) * K;

    for (int kt = 0; kt < K; kt += 32) {
        AFrag a, b0f, b1f;
        // A 16x32 f16: e -> k = 16h+e
        a.h[0] = *(const v8h*)(arow + kt + 16 * h);
        a.h[1] = *(const v8h*)(arow + kt + 16 * h + 8);
        // B 32x16 f16 (column-major per lane): e -> k = 16h+e
        b0f.h[0] = *(const v8h*)(b0p + kt + 16 * h);
        b0f.h[1] = *(const v8h*)(b0p + kt + 16 * h + 8);
        b1f.h[0] = *(const v8h*)(b1p + kt + 16 * h);
        b1f.h[1] = *(const v8h*)(b1p + kt + 16 * h + 8);
        c0 = __builtin_amdgcn_wmma_f32_16x16x32_f16(false, a.v, false, b0f.v, (short)0, c0, false, false);
        c1 = __builtin_amdgcn_wmma_f32_16x16x32_f16(false, a.v, false, b1f.v, (short)0, c1, false, false);
    }

    float bi0 = bias[n0 + r];
    float bi1 = bias[n0 + 16 + r];
#pragma unroll
    for (int rr = 0; rr < 8; ++rr) {
        int m = m0 + rr + 8 * h;   // C/D: VGPR rr -> M = rr (lanes 0-15) / rr+8 (lanes 16-31)
        float v0 = fmaxf(c0[rr] + bi0, 0.0f);
        float v1 = fmaxf(c1[rr] + bi1, 0.0f);
        if (out_f) {
            out_f[(size_t)m * N + n0 + r]      = v0;
            out_f[(size_t)m * N + n0 + 16 + r] = v1;
        } else {
            out_h[(size_t)m * N + n0 + r]      = (_Float16)v0;
            out_h[(size_t)m * N + n0 + 16 + r] = (_Float16)v1;
        }
    }
}

// ---------------------------------------------------------------------------
// Top-K nearest neighbor (2D) — one wave per query point.
// Per-lane register-resident sorted list, merged through LDS with a wave32
// butterfly (key = d2_bits<<32 | idx : sorts by distance, ties by index).
// LSE mode: no window, valid if d2 <= thresh.  SA mode: +1e10 outside window,
// valid if d2 < thresh.  Invalid -> nearest index.
// ---------------------------------------------------------------------------
template <int KS, bool LSE>
__global__ void knn_kernel(const float* __restrict__ qxyz, const float* __restrict__ sxyz,
                           int* __restrict__ idx_out, int nq, int ns, float half_w, float thresh)
{
    __shared__ unsigned long long keys[32 * KS];
    __shared__ unsigned long long sel[KS];
    const unsigned long long KMAX = ~0ull;
    int lane = threadIdx.x;
    int q = blockIdx.x;           // over B*nq
    int b = q / nq;
    const float* sb = sxyz + (size_t)b * ns * 3;
    float qx = qxyz[(size_t)q * 3 + 0];
    float qy = qxyz[(size_t)q * 3 + 1];

    unsigned long long list[KS];
#pragma unroll
    for (int t = 0; t < KS; ++t) list[t] = KMAX;

    for (int j = lane; j < ns; j += 32) {
        float dx = sb[(size_t)j * 3 + 0] - qx;
        float dy = sb[(size_t)j * 3 + 1] - qy;
        float d2 = dx * dx + dy * dy;
        if (!LSE) { if (fabsf(dx) > half_w || fabsf(dy) > half_w) d2 += 1.0e10f; }
        unsigned long long key = ((unsigned long long)__float_as_uint(d2) << 32) | (unsigned int)j;
        if (key < list[KS - 1]) {
            list[KS - 1] = key;
#pragma unroll
            for (int s = KS - 1; s > 0; --s) {
                if (list[s] < list[s - 1]) {
                    unsigned long long tmp = list[s]; list[s] = list[s - 1]; list[s - 1] = tmp;
                }
            }
        }
    }
#pragma unroll
    for (int t = 0; t < KS; ++t) keys[t * 32 + lane] = list[t];
    __syncthreads();

    for (int rsel = 0; rsel < KS; ++rsel) {
        unsigned long long bk = KMAX; int bs = 0;
        for (int t = 0; t < KS; ++t) {
            unsigned long long k2 = keys[t * 32 + lane];
            if (k2 < bk) { bk = k2; bs = t * 32 + lane; }
        }
        for (int off = 16; off; off >>= 1) {
            unsigned long long ok = __shfl_xor(bk, off, 32);
            int os = __shfl_xor(bs, off, 32);
            if (ok < bk) { bk = ok; bs = os; }
        }
        if (lane == 0) { sel[rsel] = bk; keys[bs] = KMAX; }
        __syncthreads();
    }

    if (lane < KS) {
        unsigned long long k0 = sel[0];
        unsigned long long k  = sel[lane];
        int id0 = (int)(unsigned int)k0;
        int id  = (int)(unsigned int)k;
        float d = __uint_as_float((unsigned int)(k >> 32));
        bool ok = LSE ? (d <= thresh) : (d < thresh);
        idx_out[(size_t)q * KS + lane] = ok ? id : id0;
    }
}

// ---------------------------------------------------------------------------
// Farthest point sampling (2D) — one workgroup per batch, min-dist array in
// LDS, block-wide argmax each step (ties -> smallest index, as jnp.argmax).
// ---------------------------------------------------------------------------
__global__ void fps_kernel(const float* __restrict__ xyz, int* __restrict__ idx_out, int n, int m)
{
    extern __shared__ float mind[];
    __shared__ unsigned long long warp_red[8];
    __shared__ int s_last;
    int b = blockIdx.x, tid = threadIdx.x;
    const float* base = xyz + (size_t)b * n * 3;

    for (int i = tid; i < n; i += blockDim.x) mind[i] = 1.0e10f;
    if (tid == 0) { idx_out[(size_t)b * m] = 0; s_last = 0; }
    __syncthreads();

    int last = 0;
    for (int k = 1; k < m; ++k) {
        float lx = base[(size_t)last * 3 + 0];
        float ly = base[(size_t)last * 3 + 1];
        unsigned long long best = 0;
        for (int i = tid; i < n; i += blockDim.x) {
            float dx = base[(size_t)i * 3 + 0] - lx;
            float dy = base[(size_t)i * 3 + 1] - ly;
            float d = dx * dx + dy * dy;
            float mn = fminf(mind[i], d);
            mind[i] = mn;
            unsigned long long key = ((unsigned long long)__float_as_uint(mn) << 32)
                                   | (unsigned int)(0xFFFFFFFFu - (unsigned int)i);
            if (key > best) best = key;
        }
        for (int off = 16; off; off >>= 1) {
            unsigned long long o = __shfl_xor(best, off, 32);
            if (o > best) best = o;
        }
        if ((tid & 31) == 0) warp_red[tid >> 5] = best;
        __syncthreads();
        if (tid == 0) {
            unsigned long long rmax = warp_red[0];
            int nw = blockDim.x >> 5;
            for (int w = 1; w < nw; ++w) if (warp_red[w] > rmax) rmax = warp_red[w];
            int win = (int)(0xFFFFFFFFu - (unsigned int)rmax);
            s_last = win;
            idx_out[(size_t)b * m + k] = win;
        }
        __syncthreads();
        last = s_last;
    }
}

__global__ void gather_points_kernel(const float* __restrict__ src, const int* __restrict__ idx,
                                     float* __restrict__ dst, int n, int m, int total)
{
    int t = blockIdx.x * blockDim.x + threadIdx.x;
    if (t >= total) return;
    int b = t / m;
    int j = idx[t];
    const float* s = src + (size_t)(b * n + j) * 3;
    dst[(size_t)t * 3 + 0] = s[0];
    dst[(size_t)t * 3 + 1] = s[1];
    dst[(size_t)t * 3 + 2] = s[2];
}

// ---------------------------------------------------------------------------
// LSE MLP: per point, 16 neighbors, MLP 3->16->16 relu, max over neighbors.
// feats[p] = [z, f0..f15]
// ---------------------------------------------------------------------------
__global__ void lse_mlp_kernel(const float* __restrict__ xyz, const int* __restrict__ idx,
                               const float* __restrict__ W0, const float* __restrict__ b0,
                               const float* __restrict__ W1, const float* __restrict__ b1,
                               float* __restrict__ feats, int N, int total)
{
    __shared__ float sW0[48], sb0[16], sW1[256], sb1[16];
    int tid = threadIdx.x;
    for (int i = tid; i < 48; i += blockDim.x) sW0[i] = W0[i];
    for (int i = tid; i < 16; i += blockDim.x) { sb0[i] = b0[i]; sb1[i] = b1[i]; }
    for (int i = tid; i < 256; i += blockDim.x) sW1[i] = W1[i];
    __syncthreads();

    int p = blockIdx.x * blockDim.x + tid;
    if (p >= total) return;
    int b = p / N;
    float px = xyz[(size_t)p * 3 + 0];
    float py = xyz[(size_t)p * 3 + 1];
    float pz = xyz[(size_t)p * 3 + 2];

    float fm[16];
#pragma unroll
    for (int o = 0; o < 16; ++o) fm[o] = 0.0f;   // relu outputs are >= 0

    for (int s = 0; s < 16; ++s) {
        int j = idx[(size_t)p * 16 + s];
        const float* g = xyz + (size_t)(b * N + j) * 3;
        float rx = g[0] - px, ry = g[1] - py, rz = g[2] - pz;
        float h1[16];
#pragma unroll
        for (int o = 0; o < 16; ++o)
            h1[o] = fmaxf(rx * sW0[o] + ry * sW0[16 + o] + rz * sW0[32 + o] + sb0[o], 0.0f);
#pragma unroll
        for (int o = 0; o < 16; ++o) {
            float a = sb1[o];
#pragma unroll
            for (int k = 0; k < 16; ++k) a += h1[k] * sW1[k * 16 + o];
            fm[o] = fmaxf(fm[o], fmaxf(a, 0.0f));
        }
    }
    feats[(size_t)p * 17] = pz;
#pragma unroll
    for (int o = 0; o < 16; ++o) feats[(size_t)p * 17 + 1 + o] = fm[o];
}

// ---------------------------------------------------------------------------
// Build SA GEMM input: row (b,q,s) -> [rel_xyz(3), feats(Cf), pad]  as f16
// ---------------------------------------------------------------------------
__global__ void build_sa_A_kernel(const float* __restrict__ sxyz, const float* __restrict__ sfeat,
                                  const float* __restrict__ nxyz, const int* __restrict__ idx,
                                  _Float16* __restrict__ A, int n, int npoint, int Cf, int Kpad, int total)
{
    int row = blockIdx.x * blockDim.x + threadIdx.x;
    if (row >= total) return;
    int pq = row >> 5;
    int q = pq % npoint;
    int b = pq / npoint;
    (void)q;
    int j = idx[row];
    const float* sp  = sxyz + (size_t)(b * n + j) * 3;
    const float* np_ = nxyz + (size_t)pq * 3;
    _Float16* a = A + (size_t)row * Kpad;
    a[0] = (_Float16)(sp[0] - np_[0]);
    a[1] = (_Float16)(sp[1] - np_[1]);
    a[2] = (_Float16)(sp[2] - np_[2]);
    const float* f = sfeat + (size_t)(b * n + j) * Cf;
    for (int c = 0; c < Cf; ++c) a[3 + c] = (_Float16)f[c];
    for (int k = 3 + Cf; k < Kpad; ++k) a[k] = (_Float16)0.0f;
}

// max over the 32 samples of each new point
__global__ void maxpool32_kernel(const _Float16* __restrict__ in, float* __restrict__ out,
                                 int C, int total)
{
    int t = blockIdx.x * blockDim.x + threadIdx.x;
    if (t >= total) return;
    int pq = t / C, c = t % C;
    const _Float16* p = in + (size_t)pq * 32 * C + c;
    float m = (float)p[0];
    for (int s = 1; s < 32; ++s) m = fmaxf(m, (float)p[(size_t)s * C]);
    out[t] = m;
}

// ---------------------------------------------------------------------------
// FP build: 3-NN inverse-distance interp of coarse feats + concat fine feats
// ---------------------------------------------------------------------------
__global__ void fp_build_kernel(const float* __restrict__ fxyz, const float* __restrict__ cxyz,
                                const float* __restrict__ cfeat, const float* __restrict__ ffeat,
                                _Float16* __restrict__ A, int nf, int nc, int Cc, int Cf,
                                int Kpad, int total)
{
    int t = blockIdx.x * blockDim.x + threadIdx.x;
    if (t >= total) return;
    int b = t / nf;
    float qx = fxyz[(size_t)t * 3 + 0];
    float qy = fxyz[(size_t)t * 3 + 1];
    const float* cb = cxyz + (size_t)b * nc * 3;
    const unsigned long long KMAX = ~0ull;
    unsigned long long k0 = KMAX, k1 = KMAX, k2 = KMAX;
    for (int j = 0; j < nc; ++j) {
        float dx = cb[(size_t)j * 3 + 0] - qx;
        float dy = cb[(size_t)j * 3 + 1] - qy;
        float d2 = dx * dx + dy * dy;
        unsigned long long key = ((unsigned long long)__float_as_uint(d2) << 32) | (unsigned int)j;
        if (key < k2) {
            k2 = key;
            if (k2 < k1) { unsigned long long tt = k1; k1 = k2; k2 = tt; }
            if (k1 < k0) { unsigned long long tt = k0; k0 = k1; k1 = tt; }
        }
    }
    float d0 = __uint_as_float((unsigned int)(k0 >> 32));
    float d1 = __uint_as_float((unsigned int)(k1 >> 32));
    float d2_ = __uint_as_float((unsigned int)(k2 >> 32));
    float w0 = 1.0f / fmaxf(d0, 1e-10f);
    float w1 = 1.0f / fmaxf(d1, 1e-10f);
    float w2 = 1.0f / fmaxf(d2_, 1e-10f);
    float ws = w0 + w1 + w2;
    w0 /= ws; w1 /= ws; w2 /= ws;
    int i0 = (int)(unsigned int)k0;
    int i1 = (int)(unsigned int)k1;
    int i2 = (int)(unsigned int)k2;
    const float* c0p = cfeat + (size_t)(b * nc + i0) * Cc;
    const float* c1p = cfeat + (size_t)(b * nc + i1) * Cc;
    const float* c2p = cfeat + (size_t)(b * nc + i2) * Cc;
    _Float16* a = A + (size_t)t * Kpad;
    for (int c = 0; c < Cc; ++c)
        a[c] = (_Float16)(w0 * c0p[c] + w1 * c1p[c] + w2 * c2p[c]);
    const float* fp = ffeat + (size_t)t * Cf;
    for (int c = 0; c < Cf; ++c) a[Cc + c] = (_Float16)fp[c];
    for (int k = Cc + Cf; k < Kpad; ++k) a[k] = (_Float16)0.0f;
}

// ===========================================================================
extern "C" void kernel_launch(void* const* d_in, const int* in_sizes, int n_in,
                              void* d_out, int out_size, void* d_ws, size_t ws_size,
                              hipStream_t stream)
{
    const int Bb = 2, Nn = 8192;

    auto find = [&](int sz) -> int {
        for (int i = 0; i < n_in; ++i) if (in_sizes[i] == sz) return i;
        return -1;
    };
    int i_xyz = find(49152);
    int i_lse = find(48);
    int i_sa1 = find(640);
    int i_sa2 = find(1120);
    int i_sa3 = find(4288);
    int i_sa4 = find(16768);
    int i_fp4 = find(98304);
    int i_fp3 = find(81920);
    int i_fp2 = find(73728);
    int i_fp1 = find(69888);
    if (i_xyz < 0 || i_lse < 0 || i_sa1 < 0 || i_sa2 < 0 || i_sa3 < 0 ||
        i_sa4 < 0 || i_fp4 < 0 || i_fp3 < 0 || i_fp2 < 0 || i_fp1 < 0) return;

    const float* xyz = (const float*)d_in[i_xyz];
    auto P = [&](int base, int off) { return (const float*)d_in[base + off]; };

    // ---- workspace layout --------------------------------------------------
    char* wsp = (char*)d_ws;
    size_t off = 0;
    auto alloc = [&](size_t bytes) -> void* {
        void* p = wsp + off;
        off = (off + bytes + 255) & ~(size_t)255;
        return p;
    };
    int*   idx_lse = (int*)  alloc((size_t)Bb * Nn * 16 * 4);
    float* feats0  = (float*)alloc((size_t)Bb * Nn * 17 * 4);
    int*   fps1 = (int*)alloc((size_t)Bb * 2048 * 4);
    float* l1x  = (float*)alloc((size_t)Bb * 2048 * 3 * 4);
    float* l1f  = (float*)alloc((size_t)Bb * 2048 * 32 * 4);
    int*   fps2 = (int*)alloc((size_t)Bb * 512 * 4);
    float* l2x  = (float*)alloc((size_t)Bb * 512 * 3 * 4);
    float* l2f  = (float*)alloc((size_t)Bb * 512 * 64 * 4);
    int*   fps3 = (int*)alloc((size_t)Bb * 128 * 4);
    float* l3x  = (float*)alloc((size_t)Bb * 128 * 3 * 4);
    float* l3f  = (float*)alloc((size_t)Bb * 128 * 128 * 4);
    int*   fps4 = (int*)alloc((size_t)Bb * 32 * 4);
    float* l4x  = (float*)alloc((size_t)Bb * 32 * 3 * 4);
    float* l4f  = (float*)alloc((size_t)Bb * 32 * 256 * 4);
    int* knn1 = (int*)alloc((size_t)Bb * 2048 * 32 * 4);
    int* knn2 = (int*)alloc((size_t)Bb * 512 * 32 * 4);
    int* knn3 = (int*)alloc((size_t)Bb * 128 * 32 * 4);
    int* knn4 = (int*)alloc((size_t)Bb * 32 * 32 * 4);
    float* l3f2 = (float*)alloc((size_t)Bb * 128 * 256 * 4);
    float* l2f2 = (float*)alloc((size_t)Bb * 512 * 256 * 4);
    float* l1f2 = (float*)alloc((size_t)Bb * 2048 * 256 * 4);

    // ---- GEMM weight descriptors ------------------------------------------
    struct GW { const float* W; const float* b; int K, Kpad, N; _Float16* WT; };
    GW gw[17];
    int gi = 0;
    auto addg = [&](const float* W, const float* b, int K, int Kpad, int N) {
        gw[gi].W = W; gw[gi].b = b; gw[gi].K = K; gw[gi].Kpad = Kpad; gw[gi].N = N;
        gw[gi].WT = (_Float16*)alloc((size_t)N * Kpad * 2);
        ++gi;
    };
    addg(P(i_sa1,0), P(i_sa1,1),  20,  32,  32);
    addg(P(i_sa1,2), P(i_sa1,3),  32,  32,  32);
    addg(P(i_sa1,4), P(i_sa1,5),  32,  32,  32);
    addg(P(i_sa2,0), P(i_sa2,1),  35,  64,  32);
    addg(P(i_sa2,2), P(i_sa2,3),  32,  32,  32);
    addg(P(i_sa2,4), P(i_sa2,5),  32,  32,  64);
    addg(P(i_sa3,0), P(i_sa3,1),  67,  96,  64);
    addg(P(i_sa3,2), P(i_sa3,3),  64,  64,  64);
    addg(P(i_sa3,4), P(i_sa3,5),  64,  64, 128);
    addg(P(i_sa4,0), P(i_sa4,1), 131, 160, 128);
    addg(P(i_sa4,2), P(i_sa4,3), 128, 128, 128);
    addg(P(i_sa4,4), P(i_sa4,5), 128, 128, 256);
    addg(P(i_fp4,0), P(i_fp4,1), 384, 384, 256);
    addg(P(i_fp3,0), P(i_fp3,1), 320, 320, 256);
    addg(P(i_fp2,0), P(i_fp2,1), 288, 288, 256);
    addg(P(i_fp1,0), P(i_fp1,1), 273, 288, 256);
    addg(P(i_fp1,2), P(i_fp1,3), 256, 256, 256);

    _Float16* bufA = (_Float16*)alloc((size_t)4718592 * 2);  // max A / last-layer out
    _Float16* bufB = (_Float16*)alloc((size_t)4194304 * 2);  // act1
    _Float16* bufC = (_Float16*)alloc((size_t)4194304 * 2);  // act2
    (void)ws_size; (void)out_size;

    for (int i = 0; i < 17; ++i) {
        int tot = gw[i].N * gw[i].Kpad;
        convert_wt_kernel<<<(tot + 255) / 256, 256, 0, stream>>>(
            gw[i].W, gw[i].WT, gw[i].K, gw[i].Kpad, gw[i].N, tot);
    }

    auto gemm = [&](const _Float16* A, int g, _Float16* oh, float* of, int M) {
        dim3 grid(M / 16, gw[g].N / 32);
        gemm_wmma_kernel<<<grid, 32, 0, stream>>>(A, gw[g].WT, gw[g].b, oh, of, M, gw[g].Kpad, gw[g].N);
    };

    // ---- LSE ---------------------------------------------------------------
    knn_kernel<16, true><<<Bb * Nn, 32, 0, stream>>>(xyz, xyz, idx_lse, Nn, Nn, 0.0f, 0.09f);
    {
        int tot = Bb * Nn;
        lse_mlp_kernel<<<(tot + 255) / 256, 256, 0, stream>>>(
            xyz, idx_lse, P(i_lse,0), P(i_lse,1), P(i_lse,2), P(i_lse,3), feats0, Nn, tot);
    }

    // ---- SA levels ---------------------------------------------------------
    struct SAlvl { const float* sx; const float* sf; int n, npoint; float w; int Cf; int g0;
                   float* ox; float* of; int* fidx; int* kidx; };
    SAlvl L[4] = {
        { xyz, feats0, 8192, 2048, 0.2f,  17, 0, l1x, l1f, fps1, knn1 },
        { l1x, l1f,    2048,  512, 0.4f,  32, 3, l2x, l2f, fps2, knn2 },
        { l2x, l2f,     512,  128, 0.8f,  64, 6, l3x, l3f, fps3, knn3 },
        { l3x, l3f,     128,   32, 1.6f, 128, 9, l4x, l4f, fps4, knn4 },
    };
    for (int lv = 0; lv < 4; ++lv) {
        const SAlvl& s = L[lv];
        fps_kernel<<<Bb, 256, s.n * 4, stream>>>(s.sx, s.fidx, s.n, s.npoint);
        int tq = Bb * s.npoint;
        gather_points_kernel<<<(tq + 127) / 128, 128, 0, stream>>>(s.sx, s.fidx, s.ox, s.n, s.npoint, tq);
        knn_kernel<32, false><<<tq, 32, 0, stream>>>(s.ox, s.sx, s.kidx, s.npoint, s.n, s.w, 1.0e9f);
        int M = tq * 32;
        build_sa_A_kernel<<<(M + 127) / 128, 128, 0, stream>>>(
            s.sx, s.sf, s.ox, s.kidx, bufA, s.n, s.npoint, s.Cf, gw[s.g0].Kpad, M);
        gemm(bufA, s.g0,     bufB, nullptr, M);
        gemm(bufB, s.g0 + 1, bufC, nullptr, M);
        gemm(bufC, s.g0 + 2, bufA, nullptr, M);
        int C3 = gw[s.g0 + 2].N;
        int tot = tq * C3;
        maxpool32_kernel<<<(tot + 255) / 256, 256, 0, stream>>>(bufA, s.of, C3, tot);
    }

    // ---- FP levels ---------------------------------------------------------
    struct FPl { const float* fx; const float* ff; int Cf, nf; const float* cx; const float* cf;
                 int nc, g0, nlayers; float* out; };
    FPl F[4] = {
        { l3x, l3f,  128,  128, l4x, l4f,    32, 12, 1, l3f2 },
        { l2x, l2f,   64,  512, l3x, l3f2,  128, 13, 1, l2f2 },
        { l1x, l1f,   32, 2048, l2x, l2f2,  512, 14, 1, l1f2 },
        { xyz, feats0, 17, 8192, l1x, l1f2, 2048, 15, 2, (float*)d_out },
    };
    for (int lv = 0; lv < 4; ++lv) {
        const FPl& f = F[lv];
        int t = Bb * f.nf;
        fp_build_kernel<<<(t + 127) / 128, 128, 0, stream>>>(
            f.fx, f.cx, f.cf, f.ff, bufA, f.nf, f.nc, 256, f.Cf, gw[f.g0].Kpad, t);
        if (f.nlayers == 1) {
            gemm(bufA, f.g0, nullptr, f.out, t);
        } else {
            gemm(bufA, f.g0,     bufB, nullptr, t);
            gemm(bufB, f.g0 + 1, nullptr, f.out, t);
        }
    }
}
